// MultiheadAttentionLayer_22737556865782
// MI455X (gfx1250) — compile-verified, matrix operand order checked by reading, not run on hardware
//
#include <hip/hip_runtime.h>
#include <hip/hip_bf16.h>

// ---------------------------------------------------------------------------
// MultiheadAttentionLayer for MI455X (gfx1250, wave32, WMMA bf16).
// B=2, T=2048, D_MODEL=1024, N_HEADS=16, D_HEAD=64.
// Pipeline: f32->bf16 convert -> 3x NT-GEMM (Q,K,V proj; V stored transposed)
//           -> flash attention (online softmax) -> NT-GEMM output projection.
// GEMM uses double-buffered LDS fed by GLOBAL_LOAD_ASYNC_TO_LDS_B128 with
// partial s_wait_asynccnt for prefetch overlap. Attention stages K/V tiles
// into LDS via async loads, shared by all 8 waves.
// ---------------------------------------------------------------------------

typedef __attribute__((ext_vector_type(16))) __bf16 v16bf;
typedef __attribute__((ext_vector_type(8)))  __bf16 v8bf;
typedef __attribute__((ext_vector_type(8)))  float  v8f;
typedef int v4i_g __attribute__((vector_size(16)));   // matches builtin param

#define D_MODEL 1024
#define N_HEADS 16
#define D_HEAD  64
#define T_SEQ   2048
#define M_ROWS  4096        // B * T
#define LDA     40          // LDS row stride (elements) for 32-wide K slices

__device__ __forceinline__ __bf16 f2bf(float f) {
    union { float f; unsigned u; } v; v.f = f;
    unsigned r = v.u + 0x7FFFu + ((v.u >> 16) & 1u);   // round-to-nearest-even
    unsigned short h = (unsigned short)(r >> 16);
    __bf16 out;
    __builtin_memcpy(&out, &h, 2);
    return out;
}

// 16-byte global -> LDS async copy (ASYNCcnt) with non-async fallback.
__device__ __forceinline__ void stage_b128(__bf16* lds_dst, const __bf16* gsrc) {
#if __has_builtin(__builtin_amdgcn_global_load_async_to_lds_b128)
    __builtin_amdgcn_global_load_async_to_lds_b128(
        (__attribute__((address_space(1))) v4i_g*)(void*)gsrc,
        (__attribute__((address_space(3))) v4i_g*)(void*)lds_dst,
        0, 0);
#else
    *(v8bf*)lds_dst = *(const v8bf*)gsrc;
#endif
}

template <int N>
__device__ __forceinline__ void wait_async_cnt() {
#if __has_builtin(__builtin_amdgcn_s_wait_asynccnt)
    __builtin_amdgcn_s_wait_asynccnt(N);
#else
    asm volatile("s_wait_asynccnt %0" :: "i"(N) : "memory");
#endif
}

// Build a 16x32 / 32x16 bf16 WMMA fragment from two contiguous 16-byte chunks
// (CDNA5 16-bit A/B layout: elements 0..7 = chunk0, 8..15 = chunk1 per lane).
__device__ __forceinline__ v16bf make_frag(const __bf16* p0, const __bf16* p1) {
    v8bf lo = *(const v8bf*)p0;
    v8bf hi = *(const v8bf*)p1;
    return __builtin_shufflevector(lo, hi,
        0, 1, 2, 3, 4, 5, 6, 7, 8, 9, 10, 11, 12, 13, 14, 15);
}

// ------------------------------ convert ------------------------------------
__global__ void f32_to_bf16_kernel(const float* __restrict__ in,
                                   __bf16* __restrict__ out, int n) {
    int i = blockIdx.x * blockDim.x + threadIdx.x;
    int stride = gridDim.x * blockDim.x;
    for (; i < n; i += stride) out[i] = f2bf(in[i]);
}

// ------------------------------ NT GEMM ------------------------------------
// C[M,N] = A[M,K] * W[N,K]^T + bias[N]     (M=4096, N=1024, K=1024)
// MODE 0: bf16 out scattered to (b,h,t,d)        (Q / K projections)
// MODE 1: bf16 out scattered to (b,h,d,t)        (V projection, transposed)
// MODE 2: f32 out row-major [M,N]                (output projection)
template <int MODE>
__global__ void __launch_bounds__(256)
gemm_nt_kernel(const __bf16* __restrict__ A, const __bf16* __restrict__ W,
               const float* __restrict__ bias, void* __restrict__ outp) {
    __shared__ __align__(16) __bf16 ldsA[2][128 * LDA];
    __shared__ __align__(16) __bf16 ldsB[2][128 * LDA];

    const int Kdim = D_MODEL;
    const int NT_K = Kdim / 32;     // 32 K-slices
    const int m0 = blockIdx.x * 128;
    const int n0 = blockIdx.y * 128;
    const int tid  = threadIdx.x;
    const int lane = tid & 31;
    const int w    = tid >> 5;      // 8 waves
    const int lrow = lane & 15;
    const int half = lane >> 4;     // 0: K-chunks {0..7,16..23}, 1: {8..15,24..31}
    const int wm = w >> 2;          // 0..1 -> 64 rows each
    const int wn = w & 3;           // 0..3 -> 32 cols each

    auto stage_slice = [&](int buf, int k0) {
#pragma unroll
        for (int j = 0; j < 2; j++) {
            int chunk = tid + j * 256;          // 0..511
            int row = chunk >> 2;               // 0..127
            int cc  = (chunk & 3) * 8;          // 0,8,16,24
            stage_b128(&ldsA[buf][row * LDA + cc],
                       &A[(size_t)(m0 + row) * Kdim + k0 + cc]);
            stage_b128(&ldsB[buf][row * LDA + cc],
                       &W[(size_t)(n0 + row) * Kdim + k0 + cc]);
        }
    };

    v8f zero = {};
    v8f acc[4][2];
#pragma unroll
    for (int mt = 0; mt < 4; mt++)
#pragma unroll
        for (int nt = 0; nt < 2; nt++) acc[mt][nt] = zero;

    stage_slice(0, 0);
    for (int kt = 0; kt < NT_K; kt++) {
        const int buf = kt & 1;
        if (kt + 1 < NT_K) {
            // prefetch next slice into the other buffer (safe: trailing
            // barrier of prev iter guarantees it is no longer being read),
            // then wait until only the 4 newest (the prefetch) remain.
            stage_slice(buf ^ 1, (kt + 1) * 32);
            wait_async_cnt<4>();
        } else {
            wait_async_cnt<0>();
        }
        __syncthreads();

        v16bf afrag[4], bfrag[2];
#pragma unroll
        for (int mt = 0; mt < 4; mt++) {
            const __bf16* p = &ldsA[buf][(wm * 64 + mt * 16 + lrow) * LDA + half * 8];
            afrag[mt] = make_frag(p, p + 16);
        }
#pragma unroll
        for (int nt = 0; nt < 2; nt++) {
            const __bf16* p = &ldsB[buf][(wn * 32 + nt * 16 + lrow) * LDA + half * 8];
            bfrag[nt] = make_frag(p, p + 16);
        }
#pragma unroll
        for (int mt = 0; mt < 4; mt++)
#pragma unroll
            for (int nt = 0; nt < 2; nt++)
                acc[mt][nt] = __builtin_amdgcn_wmma_f32_16x16x32_bf16(
                    false, afrag[mt], false, bfrag[nt],
                    (short)0, acc[mt][nt], false, false);
        __syncthreads();
    }

    // epilogue: bias + store
#pragma unroll
    for (int mt = 0; mt < 4; mt++)
#pragma unroll
        for (int nt = 0; nt < 2; nt++)
#pragma unroll
            for (int s = 0; s < 8; s++) {
                int m = m0 + wm * 64 + mt * 16 + s + half * 8;
                int n = n0 + wn * 32 + nt * 16 + lrow;
                float val = acc[mt][nt][s] + bias[n];
                if (MODE == 2) {
                    ((float*)outp)[(size_t)m * D_MODEL + n] = val;
                } else {
                    int b = m >> 11, t = m & (T_SEQ - 1);
                    int h = n >> 6,  d = n & (D_HEAD - 1);
                    int bh = b * N_HEADS + h;
                    __bf16* ob = (__bf16*)outp;
                    if (MODE == 0)
                        ob[((size_t)bh * T_SEQ + t) * D_HEAD + d] = f2bf(val);
                    else
                        ob[((size_t)bh * D_HEAD + d) * T_SEQ + t] = f2bf(val);
                }
            }
}

// ---------------------------- flash attention ------------------------------
// Q,K: (BH, T, 64) bf16.  Vt: (BH, 64, T) bf16.  ctx: (BH, T, 64) bf16
// (which IS the reference's quirky reshape order for the final projection).
// Workgroup: 256 threads = 8 waves; 128 query rows per block, 16 per wave.
// K / Vt 64x64 tiles staged into LDS (async) once per key block, shared by
// all waves; P tile round-trips a per-wave LDS scratch for layout conversion.
__global__ void __launch_bounds__(256)
attn_kernel(const __bf16* __restrict__ Q, const __bf16* __restrict__ K,
            const __bf16* __restrict__ Vt, __bf16* __restrict__ ctx) {
    __shared__ __align__(16) __bf16 kTile[64 * 72];        // (t, d)
    __shared__ __align__(16) __bf16 vTile[64 * 72];        // (d, t)
    __shared__ __align__(16) __bf16 pscratch[8][16 * 72];  // per-wave P tile

    const int bh   = blockIdx.y;
    const int q0   = blockIdx.x * 128;
    const int tid  = threadIdx.x;
    const int w    = tid >> 5;
    const int lane = tid & 31;
    const int lrow = lane & 15;
    const int half = lane >> 4;
    const int qw   = q0 + w * 16;

    const __bf16* Qb = Q  + (size_t)bh * T_SEQ * D_HEAD;
    const __bf16* Kb = K  + (size_t)bh * T_SEQ * D_HEAD;
    const __bf16* Vb = Vt + (size_t)bh * D_HEAD * T_SEQ;

    // Q fragments for this wave's 16 rows (K-dim = d_head = 64 -> 2 chunks)
    v16bf qa[2];
#pragma unroll
    for (int kk = 0; kk < 2; kk++) {
        const __bf16* rp = Qb + (size_t)(qw + lrow) * D_HEAD + kk * 32 + half * 8;
        qa[kk] = make_frag(rp, rp + 16);
    }

    const float NEG_INF = -__builtin_inff();
    float mrun[8], lrun[8];
    v8f zero = {};
    v8f oacc[4];
#pragma unroll
    for (int s = 0; s < 8; s++) { mrun[s] = NEG_INF; lrun[s] = 0.f; }
#pragma unroll
    for (int nt = 0; nt < 4; nt++) oacc[nt] = zero;

    __bf16* ps = &pscratch[w][0];

    for (int t0 = 0; t0 < T_SEQ; t0 += 64) {
        // ---- async stage K (64 keys x 64 d) and Vt (64 d x 64 t) tiles ----
#pragma unroll
        for (int j = 0; j < 2; j++) {
            int chunk = tid + j * 256;       // 0..511
            int row = chunk >> 3;            // 0..63
            int cc  = (chunk & 7) * 8;       // 0..56
            stage_b128(&kTile[row * 72 + cc],
                       &Kb[(size_t)(t0 + row) * D_HEAD + cc]);
            stage_b128(&vTile[row * 72 + cc],
                       &Vb[(size_t)row * T_SEQ + t0 + cc]);
        }
        wait_async_cnt<0>();
        __syncthreads();

        // ---- S = Q K^T for 16 x 64 tile ----
        v8f sfrag[4];
#pragma unroll
        for (int nt = 0; nt < 4; nt++) {
            v8f sacc = zero;
#pragma unroll
            for (int kk = 0; kk < 2; kk++) {
                const __bf16* rp = &kTile[(nt * 16 + lrow) * 72 + kk * 32 + half * 8];
                v16bf kf = make_frag(rp, rp + 16);
                sacc = __builtin_amdgcn_wmma_f32_16x16x32_bf16(
                    false, qa[kk], false, kf, (short)0, sacc, false, false);
            }
            sfrag[nt] = sacc;
        }

        // ---- online softmax: row max across 64 keys ----
        float mnew[8], scale[8];
#pragma unroll
        for (int s = 0; s < 8; s++) {
            float mx = NEG_INF;
#pragma unroll
            for (int nt = 0; nt < 4; nt++)
                mx = fmaxf(mx, sfrag[nt][s] * 0.125f);
#pragma unroll
            for (int off = 8; off >= 1; off >>= 1)
                mx = fmaxf(mx, __shfl_xor(mx, off, 32));
            mnew[s]  = fmaxf(mrun[s], mx);
            scale[s] = __expf(mrun[s] - mnew[s]);
            mrun[s]  = mnew[s];
        }

        // ---- P = exp(S/8 - m); stash bf16 P to LDS; row sums ----
        float rowsum[8];
#pragma unroll
        for (int s = 0; s < 8; s++) rowsum[s] = 0.f;
#pragma unroll
        for (int nt = 0; nt < 4; nt++)
#pragma unroll
            for (int s = 0; s < 8; s++) {
                float p = __expf(sfrag[nt][s] * 0.125f - mnew[s]);
                rowsum[s] += p;
                int row = s + half * 8;          // C-layout row
                int col = nt * 16 + lrow;        // C-layout col
                ps[row * 72 + col] = f2bf(p);
            }
#pragma unroll
        for (int s = 0; s < 8; s++) {
            float rs = rowsum[s];
#pragma unroll
            for (int off = 8; off >= 1; off >>= 1)
                rs += __shfl_xor(rs, off, 32);
            lrun[s] = lrun[s] * scale[s] + rs;
        }

        // ---- rescale running O ----
#pragma unroll
        for (int nt = 0; nt < 4; nt++)
#pragma unroll
            for (int s = 0; s < 8; s++)
                oacc[nt][s] *= scale[s];

        // ---- O += P @ V  (contraction over 64 keys, N = d_head) ----
#pragma unroll
        for (int kkt = 0; kkt < 2; kkt++) {
            const __bf16* pp = ps + lrow * 72 + kkt * 32 + half * 8;
            v16bf pa = make_frag(pp, pp + 16);
#pragma unroll
            for (int ntd = 0; ntd < 4; ntd++) {
                const __bf16* vp = &vTile[(ntd * 16 + lrow) * 72 + kkt * 32 + half * 8];
                v16bf vf = make_frag(vp, vp + 16);
                oacc[ntd] = __builtin_amdgcn_wmma_f32_16x16x32_bf16(
                    false, pa, false, vf, (short)0, oacc[ntd], false, false);
            }
        }
        __syncthreads();   // protect kTile/vTile before next block's staging
    }

    // ---- normalize and store (bh, t, d) bf16 ----
#pragma unroll
    for (int s = 0; s < 8; s++) {
        float inv = 1.0f / lrun[s];
        int row = qw + s + half * 8;
#pragma unroll
        for (int nt = 0; nt < 4; nt++) {
            int d = nt * 16 + lrow;
            ctx[((size_t)bh * T_SEQ + row) * D_HEAD + d] = f2bf(oacc[nt][s] * inv);
        }
    }
}

// ------------------------------ launcher -----------------------------------
extern "C" void kernel_launch(void* const* d_in, const int* in_sizes, int n_in,
                              void* d_out, int out_size, void* d_ws, size_t ws_size,
                              hipStream_t stream) {
    const float* query = (const float*)d_in[0];
    const float* key   = (const float*)d_in[1];
    const float* value = (const float*)d_in[2];
    const float* Wq    = (const float*)d_in[3];
    const float* bq    = (const float*)d_in[4];
    const float* Wk    = (const float*)d_in[5];
    const float* bk    = (const float*)d_in[6];
    const float* Wv    = (const float*)d_in[7];
    const float* bv    = (const float*)d_in[8];
    const float* Wh    = (const float*)d_in[9];
    const float* bh    = (const float*)d_in[10];

    const size_t MT = (size_t)M_ROWS * D_MODEL;   // 4M activation elements
    const size_t WT = (size_t)D_MODEL * D_MODEL;  // 1M weight elements

    __bf16* xq  = (__bf16*)d_ws;      // bf16 copies of inputs / weights
    __bf16* xk  = xq  + MT;
    __bf16* xv  = xk  + MT;
    __bf16* wq  = xv  + MT;
    __bf16* wk  = wq  + WT;
    __bf16* wv  = wk  + WT;
    __bf16* wh  = wv  + WT;
    __bf16* Qb  = wh  + WT;           // (BH, T, 64)
    __bf16* Kb  = Qb  + MT;           // (BH, T, 64)
    __bf16* Vtb = Kb  + MT;           // (BH, 64, T)
    __bf16* ctx = Vtb + MT;           // (BH, T, 64) == quirky (B,T,D) order

    f32_to_bf16_kernel<<<1024, 256, 0, stream>>>(query, xq, (int)MT);
    f32_to_bf16_kernel<<<1024, 256, 0, stream>>>(key,   xk, (int)MT);
    f32_to_bf16_kernel<<<1024, 256, 0, stream>>>(value, xv, (int)MT);
    f32_to_bf16_kernel<<<512,  256, 0, stream>>>(Wq,    wq, (int)WT);
    f32_to_bf16_kernel<<<512,  256, 0, stream>>>(Wk,    wk, (int)WT);
    f32_to_bf16_kernel<<<512,  256, 0, stream>>>(Wv,    wv, (int)WT);
    f32_to_bf16_kernel<<<512,  256, 0, stream>>>(Wh,    wh, (int)WT);

    dim3 gg(M_ROWS / 128, D_MODEL / 128);   // 32 x 8
    gemm_nt_kernel<0><<<gg, 256, 0, stream>>>(xq, wq, bq, Qb);
    gemm_nt_kernel<0><<<gg, 256, 0, stream>>>(xk, wk, bk, Kb);
    gemm_nt_kernel<1><<<gg, 256, 0, stream>>>(xv, wv, bv, Vtb);

    dim3 ga(T_SEQ / 128, 2 * N_HEADS);      // 16 x 32
    attn_kernel<<<ga, 256, 0, stream>>>(Qb, Kb, Vtb, ctx);

    gemm_nt_kernel<2><<<gg, 256, 0, stream>>>(ctx, wh, bh, d_out);
}
